// ONLSTMCell_747324310359
// MI455X (gfx1250) — matrix operand order repaired and checked by reading, not compiled, not run on hardware
//
#include <hip/hip_runtime.h>
#include <hip/hip_bf16.h>
#include <math.h>
#include <stdint.h>

#define BATCH   16384
#define DIN     1024
#define HIDDEN  1024
#define KDIM    2048          /* DIN + HIDDEN (concatenated GEMM) */
#define CS      128           /* chunk size */
#define GDIM    4352          /* 2*CS + 4*HIDDEN */
#define DC      0.2f

typedef __attribute__((ext_vector_type(16))) __bf16 v16bf;
typedef __attribute__((ext_vector_type(8)))  __bf16 v8bf;
typedef __attribute__((ext_vector_type(4)))  __bf16 v4bf;
typedef __attribute__((ext_vector_type(8)))  float  v8f;

union BF16x16 { v16bf v; v8bf h[2]; };

/* ---- CDNA5 async global->LDS (ASYNCcnt-tracked), via inline asm ----------- */

__device__ __forceinline__ void async_ld_b128(const void* lds_dst, const void* gaddr)
{
    uint32_t lo = (uint32_t)(uintptr_t)lds_dst;       /* flat low32 == LDS offset */
    uint64_t ga = (uint64_t)(uintptr_t)gaddr;
    asm volatile("global_load_async_to_lds_b128 %0, %1, off"
                 :: "v"(lo), "v"(ga) : "memory");
}

__device__ __forceinline__ void wait_async0()
{
    asm volatile("s_wait_asynccnt 0x0" ::: "memory");
}

/* ---------------- pack kernels: fp32 -> bf16, fold dropconnect ------------- */

__global__ __launch_bounds__(256) void pack_A_kernel(
    const float* __restrict__ x, const float* __restrict__ h,
    __bf16* __restrict__ dst)
{
    size_t i = ((size_t)blockIdx.x * 256 + threadIdx.x) * 4;
    size_t m = i >> 11;                                /* /KDIM */
    int    c = (int)(i & (KDIM - 1));
    float4 v = (c < DIN) ? *(const float4*)&x[m * DIN + c]
                         : *(const float4*)&h[m * HIDDEN + (c - DIN)];
    v4bf o;
    o.x = (__bf16)v.x; o.y = (__bf16)v.y; o.z = (__bf16)v.z; o.w = (__bf16)v.w;
    *(v4bf*)&dst[i] = o;
}

__global__ __launch_bounds__(256) void pack_W_kernel(
    const float* __restrict__ wih, const float* __restrict__ whh,
    __bf16* __restrict__ dst)
{
    size_t i = ((size_t)blockIdx.x * 256 + threadIdx.x) * 4;
    size_t n = i >> 11;
    int    c = (int)(i & (KDIM - 1));
    float4 v; float s;
    if (c < DIN) { v = *(const float4*)&wih[n * DIN + c];            s = 1.0f; }
    else         { v = *(const float4*)&whh[n * HIDDEN + (c - DIN)]; s = 1.0f - DC; }
    v4bf o;
    o.x = (__bf16)(v.x * s); o.y = (__bf16)(v.y * s);
    o.z = (__bf16)(v.z * s); o.w = (__bf16)(v.w * s);
    *(v4bf*)&dst[i] = o;
}

__global__ __launch_bounds__(256) void bias_kernel(
    const float* __restrict__ bih, const float* __restrict__ bhh,
    float* __restrict__ bias)
{
    int i = blockIdx.x * 256 + threadIdx.x;
    if (i < GDIM) bias[i] = bih[i] + bhh[i];
}

/* ---------------- WMMA GEMM: gates = A(bf16) x Wc(bf16)^T + bias -----------
   256x64 block tile, KT=64, double-buffered LDS filled by async loads.
   Each wave: 32x64 strip = 8 accumulators; 8 WMMAs per 12 ds_load_b128.     */

#define MT  256
#define NT  64
#define KT  64
#define LDT 72   /* padded LDS row stride (bf16 elems); 144B = 9*16B aligned */

__global__ __launch_bounds__(256) void gemm_gates_kernel(
    const __bf16* __restrict__ A,     /* [BATCH][KDIM] row-major */
    const __bf16* __restrict__ Wc,    /* [GDIM ][KDIM] row-major */
    const float*  __restrict__ bias,  /* [GDIM] */
    float*        __restrict__ gates) /* [BATCH][GDIM] */
{
    __shared__ __bf16 smem[2 * MT * LDT + 2 * NT * LDT];   /* 92,160 B */
    __bf16* const bufA0 = smem;
    __bf16* const bufA1 = smem + MT * LDT;
    __bf16* const bufW0 = smem + 2 * MT * LDT;
    __bf16* const bufW1 = smem + 2 * MT * LDT + NT * LDT;

    const int tid  = threadIdx.x;
    const int wv   = tid >> 5;
    const int lane = tid & 31;
    const int hi   = lane >> 4;       /* 0 | 1 : which half-wave */
    const int l16  = lane & 15;
    const size_t m0 = (size_t)blockIdx.y * MT;
    const size_t n0 = (size_t)blockIdx.x * NT;

    /* this thread's fixed slice of the cooperative tile fill (16B chunks) */
    int ar[8], ac[8], wr[2], wc[2];
    #pragma unroll
    for (int i = 0; i < 8; ++i) {
        int c = tid + i * 256; ar[i] = c >> 3; ac[i] = (c & 7) << 3;
    }
    #pragma unroll
    for (int i = 0; i < 2; ++i) {
        int c = tid + i * 256; wr[i] = c >> 3; wc[i] = (c & 7) << 3;
    }

    v8f acc[8] = {};   /* acc[nb*2+ms] : N-subtile nb, M-subtile ms */

    /* prologue: fill buffer 0 for k0 = 0 */
    #pragma unroll
    for (int i = 0; i < 8; ++i)
        async_ld_b128(&bufA0[ar[i] * LDT + ac[i]], &A[(m0 + ar[i]) * KDIM + ac[i]]);
    #pragma unroll
    for (int i = 0; i < 2; ++i)
        async_ld_b128(&bufW0[wr[i] * LDT + wc[i]], &Wc[(n0 + wr[i]) * KDIM + wc[i]]);
    wait_async0();
    __syncthreads();

    int cur = 0;
    for (int k0 = 0; k0 < KDIM; k0 += KT) {
        __bf16* const cA = cur ? bufA1 : bufA0;
        __bf16* const cW = cur ? bufW1 : bufW0;
        __bf16* const nA = cur ? bufA0 : bufA1;
        __bf16* const nW = cur ? bufW0 : bufW1;

        /* prefetch next K-slice into the other buffer (async, no VGPR staging) */
        if (k0 + KT < KDIM) {
            const int kn = k0 + KT;
            #pragma unroll
            for (int i = 0; i < 8; ++i)
                async_ld_b128(&nA[ar[i] * LDT + ac[i]],
                              &A[(m0 + ar[i]) * KDIM + kn + ac[i]]);
            #pragma unroll
            for (int i = 0; i < 2; ++i)
                async_ld_b128(&nW[wr[i] * LDT + wc[i]],
                              &Wc[(n0 + wr[i]) * KDIM + kn + wc[i]]);
        }

        /* compute on current buffer: 2 K-steps x (2 M x 4 N) = 16 WMMAs */
        #pragma unroll
        for (int ks = 0; ks < KT; ks += 32) {
            BF16x16 af[2];
            #pragma unroll
            for (int ms = 0; ms < 2; ++ms) {
                const int arow = (wv * 32 + ms * 16 + l16) * LDT + ks + hi * 8;
                af[ms].h[0] = *(const v8bf*)&cA[arow];
                af[ms].h[1] = *(const v8bf*)&cA[arow + 16];
            }
            BF16x16 bfr[4];
            #pragma unroll
            for (int nb = 0; nb < 4; ++nb) {
                const int brow = (nb * 16 + l16) * LDT + ks + hi * 8;
                bfr[nb].h[0] = *(const v8bf*)&cW[brow];
                bfr[nb].h[1] = *(const v8bf*)&cW[brow + 16];
            }
            #pragma unroll
            for (int nb = 0; nb < 4; ++nb) {
                #pragma unroll
                for (int ms = 0; ms < 2; ++ms) {
                    acc[nb * 2 + ms] = __builtin_amdgcn_wmma_f32_16x16x32_bf16(
                        false, af[ms].v, false, bfr[nb].v, (short)0,
                        acc[nb * 2 + ms], false, false);
                }
            }
        }

        wait_async0();       /* own async fills of next buffer complete      */
        __syncthreads();     /* everyone's fills done + reads of cur retired */
        cur ^= 1;
    }

    /* D layout: VGPR g -> row g + 8*hi, col = lane%16 */
    #pragma unroll
    for (int nb = 0; nb < 4; ++nb) {
        const size_t col = n0 + nb * 16 + l16;
        const float  bv  = bias[col];
        #pragma unroll
        for (int ms = 0; ms < 2; ++ms) {
            const size_t row0 = m0 + (size_t)wv * 32 + ms * 16 + hi * 8;
            #pragma unroll
            for (int g = 0; g < 8; ++g)
                gates[(row0 + g) * GDIM + col] = acc[nb * 2 + ms][g] + bv;
        }
    }
}

/* ---------------- epilogue: cumsoftmax + ON-LSTM cell update --------------- */

__device__ __forceinline__ float sigmoidf_(float x) {
    return 1.0f / (1.0f + __expf(-x));
}

__global__ __launch_bounds__(128) void onlstm_epilogue_kernel(
    const float* __restrict__ gates,
    const float* __restrict__ c0,
    float* __restrict__ h1,
    float* __restrict__ c1)
{
    __shared__ float2 s2[CS];
    const int b = blockIdx.x;
    const int j = threadIdx.x;
    const float* g = gates + (size_t)b * GDIM;

    float ci = g[j];          /* cingate logits     */
    float cf = g[CS + j];     /* cforgetgate logits */

    /* max reduce (both channels) */
    s2[j] = make_float2(ci, cf);
    __syncthreads();
    for (int off = 64; off > 0; off >>= 1) {
        if (j < off) {
            float2 a = s2[j], o = s2[j + off];
            s2[j] = make_float2(fmaxf(a.x, o.x), fmaxf(a.y, o.y));
        }
        __syncthreads();
    }
    float2 mx = s2[0];
    __syncthreads();

    float ei = __expf(ci - mx.x);
    float ef = __expf(cf - mx.y);

    /* sum reduce */
    s2[j] = make_float2(ei, ef);
    __syncthreads();
    for (int off = 64; off > 0; off >>= 1) {
        if (j < off) {
            float2 a = s2[j], o = s2[j + off];
            s2[j] = make_float2(a.x + o.x, a.y + o.y);
        }
        __syncthreads();
    }
    float2 sm = s2[0];
    __syncthreads();

    /* inclusive Hillis-Steele scan */
    s2[j] = make_float2(ei, ef);
    __syncthreads();
    for (int off = 1; off < CS; off <<= 1) {
        float2 add = make_float2(0.f, 0.f);
        if (j >= off) add = s2[j - off];
        __syncthreads();
        if (j >= off) {
            float2 a = s2[j];
            s2[j] = make_float2(a.x + add.x, a.y + add.y);
        }
        __syncthreads();
    }
    const float cing = 1.0f - s2[j].x / sm.x;   /* 1 - cumsoftmax(cin)  */
    const float cfog = s2[j].y / sm.y;          /* cumsoftmax(cforget)  */

    const float ov  = cfog * cing;
    const float fsh = cfog - ov;
    const float ish = cing - ov;

    const size_t o8 = (size_t)j * 8;
    float igv[8], fgv[8], ogv[8], cgv[8], c0v[8], h1v[8], c1v[8];
    *(float4*)&igv[0] = *(const float4*)&g[2*CS            + o8];
    *(float4*)&igv[4] = *(const float4*)&g[2*CS        + 4 + o8];
    *(float4*)&fgv[0] = *(const float4*)&g[2*CS +   HIDDEN + o8];
    *(float4*)&fgv[4] = *(const float4*)&g[2*CS+4 + HIDDEN + o8];
    *(float4*)&ogv[0] = *(const float4*)&g[2*CS + 2*HIDDEN + o8];
    *(float4*)&ogv[4] = *(const float4*)&g[2*CS+4+2*HIDDEN + o8];
    *(float4*)&cgv[0] = *(const float4*)&g[2*CS + 3*HIDDEN + o8];
    *(float4*)&cgv[4] = *(const float4*)&g[2*CS+4+3*HIDDEN + o8];
    *(float4*)&c0v[0] = *(const float4*)&c0[(size_t)b*HIDDEN     + o8];
    *(float4*)&c0v[4] = *(const float4*)&c0[(size_t)b*HIDDEN + 4 + o8];

    #pragma unroll
    for (int k = 0; k < 8; ++k) {
        float i_ = sigmoidf_(igv[k]);
        float f_ = sigmoidf_(fgv[k]);
        float o_ = sigmoidf_(ogv[k]);
        float ce = tanhf(cgv[k]);
        float fg = f_ * ov + fsh;
        float ig = i_ * ov + ish;
        float cc = fg * c0v[k] + ig * ce;
        c1v[k] = cc;
        h1v[k] = o_ * tanhf(cc);
    }

    *(float4*)&h1[(size_t)b*HIDDEN     + o8] = *(float4*)&h1v[0];
    *(float4*)&h1[(size_t)b*HIDDEN + 4 + o8] = *(float4*)&h1v[4];
    *(float4*)&c1[(size_t)b*HIDDEN     + o8] = *(float4*)&c1v[0];
    *(float4*)&c1[(size_t)b*HIDDEN + 4 + o8] = *(float4*)&c1v[4];
}

/* ---------------- launcher -------------------------------------------------- */

extern "C" void kernel_launch(void* const* d_in, const int* in_sizes, int n_in,
                              void* d_out, int out_size, void* d_ws, size_t ws_size,
                              hipStream_t stream)
{
    (void)in_sizes; (void)n_in; (void)out_size; (void)ws_size;

    const float* inputs = (const float*)d_in[0];
    const float* h0     = (const float*)d_in[1];
    const float* c0     = (const float*)d_in[2];
    const float* W_ih   = (const float*)d_in[3];
    const float* b_ih   = (const float*)d_in[4];
    const float* W_hh   = (const float*)d_in[5];
    const float* b_hh   = (const float*)d_in[6];

    /* workspace layout (all offsets 256B-aligned):
       Abf  : BATCH*KDIM bf16   =  67,108,864 B
       Wbf  : GDIM *KDIM bf16   =  17,825,792 B
       bias : GDIM f32 (32KB slot)
       gates: BATCH*GDIM f32    = 285,212,672 B          (~353 MB total) */
    char* ws = (char*)d_ws;
    const size_t ABF = (size_t)BATCH * KDIM * 2;
    const size_t WBF = (size_t)GDIM  * KDIM * 2;
    __bf16* Abf  = (__bf16*)ws;
    __bf16* Wbf  = (__bf16*)(ws + ABF);
    float*  bias = (float*) (ws + ABF + WBF);
    float*  gates= (float*) (ws + ABF + WBF + 32768);

    float* h1 = (float*)d_out;
    float* c1 = (float*)d_out + (size_t)BATCH * HIDDEN;

    pack_A_kernel<<<(BATCH * KDIM / 4) / 256, 256, 0, stream>>>(inputs, h0, Abf);
    pack_W_kernel<<<(GDIM  * KDIM / 4) / 256, 256, 0, stream>>>(W_ih, W_hh, Wbf);
    bias_kernel  <<<(GDIM + 255) / 256,       256, 0, stream>>>(b_ih, b_hh, bias);

    dim3 grid(GDIM / NT, BATCH / MT);     /* 68 x 64 blocks */
    gemm_gates_kernel<<<grid, 256, 0, stream>>>(Abf, Wbf, bias, gates);

    onlstm_epilogue_kernel<<<BATCH, CS, 0, stream>>>(gates, c0, h1, c1);
}
